// DCNN_60309930770881
// MI455X (gfx1250) — matrix operand order repaired and checked by reading, not compiled, not run on hardware
//
#include <hip/hip_runtime.h>
#include <hip/hip_bf16.h>
#include <math.h>

// ---------------------------------------------------------------------------
// DCNN forward for MI455X (gfx1250, wave32, WMMA)
//   k1: embedding gather + depthwise conv1 (K=7) + fold + kmax131 + tanh (VALU)
//   k2: grouped conv2 (K=5) + fold as ONE K=96 GEMM per row-pair using
//       v_wmma_f32_16x16x32_f16; inputs staged via global_load_async_to_lds,
//       im2col-transposed in LDS so B-fragments are single 32B ds loads.
//   k3: FC 512x1024 @ 1024x6 via WMMA, vectorized fragment loads.
// Intermediates f16 (tanh-bounded) to halve HBM traffic.
// ---------------------------------------------------------------------------

typedef _Float16 half_t;
typedef half_t v16h __attribute__((ext_vector_type(16)));
typedef half_t v8h  __attribute__((ext_vector_type(8)));
typedef float  v8f  __attribute__((ext_vector_type(8)));

#define BB    512
#define SS    256
#define DD    64
#define S1    262   // conv1 out length (256 + 6)
#define KMAX1 131   // kmax after layer 1
#define H1STR 132   // h1 row stride in halves (dword aligned rows for async b32)
#define S2    135   // conv2 out length (131 + 4)
#define CP    264   // padded conv1-row length in LDS (k1)
#define XC    152   // x_lds cols (k2): 4 left-pad + 131 data + pad (>=148)
#define OC    144   // o_lds cols (k2): 9 tiles * 16
#define KK    96    // combined (g0|g1|zero-pad) reduction length

// ======================= Kernel 1: embed + conv1 + fold + kmax131 + tanh ====
__global__ __launch_bounds__(256) void k1_embed_conv1(
    const int* __restrict__ x, const float* __restrict__ emb,
    const float* __restrict__ w1, const float* __restrict__ b1,
    half_t* __restrict__ h1)
{
  extern __shared__ char smem[];
  float*  in_lds   = (float*)smem;                       // [64][257]
  half_t* conv_lds = (half_t*)(smem + 64 * 257 * 4);     // [256][CP]
  const int b   = blockIdx.x;
  const int tid = threadIdx.x;

  for (int i = tid; i < SS * DD; i += 256) {
    int s = i >> 6, d = i & 63;
    long row = (long)x[b * SS + s];
    in_lds[d * 257 + s] = emb[row * DD + d];
  }
  __syncthreads();

  const int r = tid >> 3, f = tid & 7;
  const int ca = 16 * r + f, cb = 16 * r + 8 + f;
  float wa[7], wb[7];
#pragma unroll
  for (int k = 0; k < 7; ++k) { wa[k] = w1[ca * 7 + k]; wb[k] = w1[cb * 7 + k]; }
  const float bias = b1[ca] + b1[cb];
  const float* ra = &in_lds[(2 * r) * 257];
  const float* rb = &in_lds[(2 * r + 1) * 257];

  float vmin = 1.0e30f, vmax = -1.0e30f;
  for (int so = 0; so < S1; ++so) {
    float acc = bias;
#pragma unroll
    for (int k = 0; k < 7; ++k) {
      int si = so + k - 6;
      if (si >= 0 && si < SS) acc += wa[k] * ra[si] + wb[k] * rb[si];
    }
    half_t hv = (half_t)acc;
    conv_lds[tid * CP + so] = hv;
    float fv = (float)hv;
    vmin = fminf(vmin, fv);
    vmax = fmaxf(vmax, fv);
  }

  // order-preserving top-131 via kth-largest bisection
  const half_t* rowp = &conv_lds[tid * CP];
  float lo = vmin - 1.0f, hi = vmax;
  for (int it = 0; it < 22; ++it) {
    float mid = 0.5f * (lo + hi);
    int cnt = 0;
    for (int so = 0; so < S1; ++so) cnt += ((float)rowp[so] > mid);
    if (cnt > KMAX1) lo = mid; else hi = mid;
  }
  int c1 = 0;
  for (int so = 0; so < S1; ++so) c1 += ((float)rowp[so] > hi);
  int need2 = KMAX1 - c1, used2 = 0, outn = 0;

  half_t* orow = &h1[((long)b * 256 + tid) * H1STR];
  for (int so = 0; so < S1 && outn < KMAX1; ++so) {
    float v = (float)rowp[so];
    bool take = (v > hi);
    if (!take && v > lo && used2 < need2) { take = true; ++used2; }
    if (take) orow[outn++] = (half_t)tanhf(v);
  }
  while (outn < H1STR) orow[outn++] = (half_t)0.f;  // pad incl. stride slot
}

// ======================= Kernel 2: conv2+fold as K=96 WMMA GEMM =============
// grid (512, 4), block 128 (4 waves). Wave w -> row-pair r2 = by*4 + w:
//   D[16 j][135 t] = A'[16 x 96] * X'[96 x 135] + biasfold
// where K' 0..39 = g0 (f*5+kk), 40..79 = g1, 80..95 = zero pad.
__global__ __launch_bounds__(128) void k2_conv2_wmma(
    const half_t* __restrict__ h1, const float* __restrict__ w2,
    const float* __restrict__ b2, half_t* __restrict__ h2)
{
  extern __shared__ char smem[];
  half_t* x2t   = (half_t*)smem;                             // [4][144][KK]
  float*  o_lds = (float*)(smem + 4 * 144 * KK * 2);         // [4][16][OC]
  half_t* x_lds = (half_t*)(smem + 4 * 144 * KK * 2 + 4 * 16 * OC * 4); // [64][XC]
  const int b = blockIdx.x, by = blockIdx.y;
  const int tid  = threadIdx.x;
  const int w    = tid >> 5, lane = tid & 31;
  const int hl   = lane >> 4;      // half-wave
  const int mn   = lane & 15;      // M for A/C rows, N for B/C cols

  // ---- zero staging, then async global->LDS copy of 64 h1 rows ----
  const long growbase = (long)b * 256 + by * 64;
  __builtin_prefetch(&h1[growbase * H1STR], 0, 1);
  for (int i = tid; i < 64 * XC; i += 128) x_lds[i] = (half_t)0.f;
  __syncthreads();
  // 64 rows * 66 dwords (132 halves incl. zero pad), 4224/128 = 33 full iters
  for (int i = tid; i < 64 * 66; i += 128) {
    int lr = i / 66, dw = i - lr * 66;
    const half_t* gsrc = h1 + (growbase + lr) * H1STR + dw * 2;
    unsigned int ldsb = (unsigned int)(uintptr_t)(x_lds + lr * XC + 4 + dw * 2);
    unsigned long long ga = (unsigned long long)(uintptr_t)gsrc;
    asm volatile("global_load_async_to_lds_b32 %0, %1, off"
                 :: "v"(ldsb), "v"(ga) : "memory");
  }
  asm volatile("s_wait_asynccnt 0x0" ::: "memory");
  __syncthreads();

  const int r2  = by * 4 + w;
  const int g0  = 2 * r2, g1 = g0 + 1;
  const int lr0 = w * 16;                 // local rows: g0 -> +0..7, g1 -> +8..15
  half_t* x2w = x2t + w * 144 * KK;

  // ---- build im2col-transposed X': x2w[t][K'] (K'-contiguous per column) ----
  int rowo[3], qo[3];  bool vld[3];
#pragma unroll
  for (int j = 0; j < 3; ++j) {
    int kk = lane + 32 * j;
    if (kk < 40)       { rowo[j] = lr0 + kk / 5;            qo[j] = kk % 5;        vld[j] = true; }
    else if (kk < 80)  { int u = kk - 40; rowo[j] = lr0 + 8 + u / 5; qo[j] = u % 5; vld[j] = true; }
    else               { rowo[j] = 0; qo[j] = 0; vld[j] = false; }
  }
  for (int t = 0; t < 144; ++t) {
#pragma unroll
    for (int j = 0; j < 3; ++j)
      x2w[t * KK + lane + 32 * j] =
          vld[j] ? x_lds[rowo[j] * XC + t + qo[j]] : (half_t)0.f;
  }

  // ---- A' fragments (3 chunks of K=32), contiguous k = 16*hl + i, same as B ----
  v16h ac[3];
#pragma unroll
  for (int c = 0; c < 3; ++c) {
#pragma unroll
    for (int i = 0; i < 16; ++i) {
      int kA = c * 32 + 16 * hl + i;
      float wv;
      if (kA < 40)      wv = w2[(g0 * 16 + mn) * 40 + kA];
      else if (kA < 80) wv = w2[(g1 * 16 + mn) * 40 + (kA - 40)];
      else              wv = 0.f;
      ac[c][i] = (half_t)wv;
    }
  }
  v8f cbias;
#pragma unroll
  for (int v = 0; v < 8; ++v) {
    int m = v + 8 * hl;
    cbias[v] = b2[g0 * 16 + m] + b2[g1 * 16 + m];
  }

  // ---- 9 column tiles, 3 WMMAs each; B frag = one 32B contiguous LDS read ----
  for (int T = 0; T < 9; ++T) {
    int t0 = T * 16;
    const half_t* xrow = x2w + (t0 + mn) * KK;
    v16h b0 = *(const v16h*)(xrow + 0 * 32 + 16 * hl);
    v16h b1 = *(const v16h*)(xrow + 1 * 32 + 16 * hl);
    v16h b2f = *(const v16h*)(xrow + 2 * 32 + 16 * hl);
    v8f c = cbias;
    c = __builtin_amdgcn_wmma_f32_16x16x32_f16(false, ac[0], false, b0,  (short)0, c, false, false);
    c = __builtin_amdgcn_wmma_f32_16x16x32_f16(false, ac[1], false, b1,  (short)0, c, false, false);
    c = __builtin_amdgcn_wmma_f32_16x16x32_f16(false, ac[2], false, b2f, (short)0, c, false, false);
#pragma unroll
    for (int v = 0; v < 8; ++v)
      o_lds[(w * 16 + v + 8 * hl) * OC + t0 + mn] = c[v];
  }
  __syncthreads();

  // ---- kmax4 (order-preserving) + tanh; lane j < 16 handles row j ----
  if (lane < 16) {
    const float NEG = -3.0e38f;
    float v0 = NEG, v1 = NEG, v2 = NEG, v3 = NEG;
    int   i0 = 0, i1 = 0, i2 = 0, i3 = 0;
    const float* rowq = &o_lds[(w * 16 + lane) * OC];
    for (int t = 0; t < S2; ++t) {
      float v = rowq[t];
      if (v > v3) {
        if (v > v0)      { v3 = v2; i3 = i2; v2 = v1; i2 = i1; v1 = v0; i1 = i0; v0 = v; i0 = t; }
        else if (v > v1) { v3 = v2; i3 = i2; v2 = v1; i2 = i1; v1 = v;  i1 = t; }
        else if (v > v2) { v3 = v2; i3 = i2; v2 = v;  i2 = t; }
        else             { v3 = v;  i3 = t; }
      }
    }
    float vv[4] = { v0, v1, v2, v3 };
    int   ii[4] = { i0, i1, i2, i3 };
#pragma unroll
    for (int a = 0; a < 3; ++a)
#pragma unroll
      for (int q = 0; q < 3 - a; ++q)
        if (ii[q] > ii[q + 1]) {
          int   ti = ii[q]; ii[q] = ii[q + 1]; ii[q + 1] = ti;
          float tv = vv[q]; vv[q] = vv[q + 1]; vv[q + 1] = tv;
        }
    half_t* outp = &h2[(long)b * 1024 + (r2 * 16 + lane) * 4];
#pragma unroll
    for (int q = 0; q < 4; ++q) outp[q] = (half_t)tanhf(vv[q]);
  }
}

// ======================= Kernel 3: FC (512x1024)@(1024x6) via WMMA ==========
__global__ __launch_bounds__(32) void k3_fc_wmma(
    const half_t* __restrict__ h2, const float* __restrict__ fc_w,
    const float* __restrict__ fc_b, float* __restrict__ out)
{
  extern __shared__ char smem[];
  half_t* wlds = (half_t*)smem;       // [16][1024]
  half_t* xlds = wlds + 16 * 1024;    // [16][1024]
  const int lane = threadIdx.x;
  const int hl = lane >> 4, mn = lane & 15;
  const int b0 = blockIdx.x * 16;

  for (int i = lane; i < 16 * 1024; i += 32) {
    int row = i >> 10, k = i & 1023;
    wlds[i] = (row < 6) ? (half_t)fc_w[row * 1024 + k] : (half_t)0.f;
    xlds[i] = h2[(long)(b0 + row) * 1024 + k];
  }
  __syncthreads();

  v8f c;
  float bias = (mn < 6) ? fc_b[mn] : 0.f;
#pragma unroll
  for (int v = 0; v < 8; ++v) c[v] = bias;

  for (int kc = 0; kc < 32; ++kc) {
    int k0 = kc * 32;
    // A: two contiguous 16B loads covering k = 16*hl .. 16*hl+15 (matches B)
    v8h lo = *(const v8h*)&xlds[mn * 1024 + k0 + 16 * hl];
    v8h hi = *(const v8h*)&xlds[mn * 1024 + k0 + 16 * hl + 8];
    v16h a = __builtin_shufflevector(lo, hi, 0,1,2,3,4,5,6,7,8,9,10,11,12,13,14,15);
    // B: one contiguous 32B load
    v16h bf = *(const v16h*)&wlds[mn * 1024 + k0 + 16 * hl];
    c = __builtin_amdgcn_wmma_f32_16x16x32_f16(false, a, false, bf, (short)0, c, false, false);
  }
  if (mn < 6) {
#pragma unroll
    for (int v = 0; v < 8; ++v)
      out[(b0 + v + 8 * hl) * 6 + mn] = c[v];
  }
}

// ======================= host launcher ======================================
extern "C" void kernel_launch(void* const* d_in, const int* in_sizes, int n_in,
                              void* d_out, int out_size, void* d_ws, size_t ws_size,
                              hipStream_t stream) {
  (void)in_sizes; (void)n_in; (void)out_size; (void)ws_size;
  const int*   x   = (const int*)d_in[0];
  const float* emb = (const float*)d_in[1];
  const float* w1  = (const float*)d_in[2];
  const float* b1  = (const float*)d_in[3];
  const float* w2  = (const float*)d_in[4];
  const float* b2  = (const float*)d_in[5];
  const float* fcw = (const float*)d_in[6];
  const float* fcb = (const float*)d_in[7];
  float* out = (float*)d_out;

  // workspace: h1 (512*256*132 f16 ~= 34.6 MB) then h2 (512*1024 f16 = 1 MB)
  half_t* h1 = (half_t*)d_ws;
  size_t h1_bytes = ((size_t)BB * 256 * H1STR * sizeof(half_t) + 255) & ~(size_t)255;
  half_t* h2 = (half_t*)((char*)d_ws + h1_bytes);

  size_t smem1 = (size_t)64 * 257 * 4 + (size_t)256 * CP * 2;          // ~196 KB
  k1_embed_conv1<<<dim3(BB), dim3(256), smem1, stream>>>(x, emb, w1, b1, h1);

  size_t smem2 = (size_t)4 * 144 * KK * 2 + (size_t)4 * 16 * OC * 4
               + (size_t)64 * XC * 2;                                  // ~163 KB
  k2_conv2_wmma<<<dim3(BB, 4), dim3(128), smem2, stream>>>(h1, w2, b2, h2);

  size_t smem3 = (size_t)2 * 16 * 1024 * 2;                            // 64 KB
  k3_fc_wmma<<<dim3(32), dim3(32), smem3, stream>>>(h2, fcw, fcb, out);
}
